// MultiHeadAttentionLayer_59012850647246
// MI455X (gfx1250) — matrix operand order repaired and checked by reading, not compiled, not run on hardware
//
#include <hip/hip_runtime.h>

typedef __bf16 bf16_t;
typedef __attribute__((ext_vector_type(16))) __bf16 v16bf;
typedef __attribute__((ext_vector_type(8)))  float  v8f;

#define IN_DIM 128
#define HD 64    // H*D
#define NH 4
#define DD 16

// ---------------- zero init (d_out accumulator + z) ----------------
__global__ void zero_kernel(float* __restrict__ out, float* __restrict__ z,
                            int nOut, int nZ) {
  int t = blockIdx.x * blockDim.x + threadIdx.x;
  if (t < nOut) out[t] = 0.0f;
  if (t < nZ)   z[t]   = 0.0f;
}

// ---- pack Wq/Wk/Wv (f32 row-major [128,64]) into bf16 WMMA B-fragments ----
// frag f = k0idx*12 + mat*4 + ct ; per lane 16 contiguous bf16 (32B).
// B-fragment k order used here (same as A): lane = (k>=16)*16 + n, element j -> k = half*16 + j
__global__ void pack_w_kernel(const float* __restrict__ Wq,
                              const float* __restrict__ Wk,
                              const float* __restrict__ Wv,
                              bf16_t* __restrict__ wp) {
  int t = blockIdx.x * blockDim.x + threadIdx.x;
  if (t >= 48 * 32) return;
  int f = t >> 5, lane = t & 31;
  int k0idx = f / 12, rem = f % 12, mat = rem >> 2, ct = rem & 3;
  const float* W = (mat == 0) ? Wq : (mat == 1) ? Wk : Wv;
  int half = lane >> 4, n = lane & 15;
  bf16_t* dstp = wp + f * 512 + lane * 16;
#pragma unroll
  for (int j = 0; j < 16; ++j) {
    int kk = k0idx * 32 + half * 16 + j;
    dstp[j] = (bf16_t)W[kk * HD + ct * 16 + n];
  }
}

// ---------------- fused QKV GEMM: one wave = 16 rows x 64 cols x 3 mats ----
__global__ void __launch_bounds__(256)
qkv_gemm_kernel(const float* __restrict__ h,
                const bf16_t* __restrict__ wp,
                const float* __restrict__ bq,
                const float* __restrict__ bk,
                const float* __restrict__ bv,
                float* __restrict__ Q, float* __restrict__ K,
                float* __restrict__ V, int nTiles) {
  int wave = threadIdx.x >> 5;
  int lane = threadIdx.x & 31;
  int tile = blockIdx.x * 8 + wave;
  if (tile >= nTiles) return;          // wave-uniform: EXEC stays all-1s
  int half = lane >> 4, m = lane & 15;
  long row = (long)tile * 16 + m;

  v8f acc[3][4] = {};                  // fp32 accumulators, 96 VGPRs

  const float* hrow = h + row * IN_DIM;

#pragma unroll
  for (int kb = 0; kb < 4; ++kb) {
    // A fragment (16-bit A 16x32): elem j -> k = half*16 + j
    const float* p0 = hrow + kb * 32 + half * 16;
    const float* p1 = p0 + 8;
    v16bf a;
#pragma unroll
    for (int j = 0; j < 8; ++j) {
      a[j]     = (bf16_t)p0[j];
      a[8 + j] = (bf16_t)p1[j];
    }
#pragma unroll
    for (int ma = 0; ma < 3; ++ma) {
#pragma unroll
      for (int c = 0; c < 4; ++c) {
        const v16bf b =
            *(const v16bf*)(wp + ((kb * 12 + ma * 4 + c) * 512 + lane * 16));
        acc[ma][c] = __builtin_amdgcn_wmma_f32_16x16x32_bf16(
            false, a, false, b, (short)0, acc[ma][c], false, false);
      }
    }
  }

  // C/D layout: VGPR r, lane -> row = half*8 + r, col = lane&15
  float* outs[3]         = {Q, K, V};
  const float* biases[3] = {bq, bk, bv};
  int n = lane & 15;
#pragma unroll
  for (int ma = 0; ma < 3; ++ma) {
#pragma unroll
    for (int c = 0; c < 4; ++c) {
#pragma unroll
      for (int r = 0; r < 8; ++r) {
        long rr  = (long)tile * 16 + half * 8 + r;
        int  col = c * 16 + n;
        outs[ma][rr * HD + col] = acc[ma][c][r] + biases[ma][col];
      }
    }
  }
}

// ---------------- edge phase: one thread per (edge, head) ----------------
__global__ void __launch_bounds__(256)
edge_kernel(const int* __restrict__ src, const int* __restrict__ dst,
            const float* __restrict__ Q, const float* __restrict__ K,
            const float* __restrict__ V, float* __restrict__ wV,
            float* __restrict__ z, int EH) {
  int t = blockIdx.x * blockDim.x + threadIdx.x;
  if (t >= EH) return;
  int e = t >> 2, hd = t & 3;
  int s = src[e], d = dst[e];
  const float4* kp = (const float4*)(K + (long)s * HD + hd * DD);
  const float4* qp = (const float4*)(Q + (long)d * HD + hd * DD);
  float dot = 0.0f;
#pragma unroll
  for (int i = 0; i < 4; ++i) {
    float4 kv = kp[i], qv = qp[i];
    dot += kv.x * qv.x + kv.y * qv.y + kv.z * qv.z + kv.w * qv.w;
  }
  float sc = __expf(fminf(fmaxf(dot * 0.25f, -5.0f), 5.0f));  // 1/sqrt(16)
  const float4* vp = (const float4*)(V + (long)s * HD + hd * DD);
  float* op = wV + (long)d * HD + hd * DD;
#pragma unroll
  for (int i = 0; i < 4; ++i) {
    float4 vv = vp[i];
    atomicAdd(op + i * 4 + 0, sc * vv.x);
    atomicAdd(op + i * 4 + 1, sc * vv.y);
    atomicAdd(op + i * 4 + 2, sc * vv.z);
    atomicAdd(op + i * 4 + 3, sc * vv.w);
  }
  atomicAdd(z + (long)d * NH + hd, sc);
}

// ---------------- normalize in place ----------------
__global__ void norm_kernel(float* __restrict__ out, const float* __restrict__ z,
                            int total) {
  int t = blockIdx.x * blockDim.x + threadIdx.x;
  if (t >= total) return;
  int node = t >> 6;
  int hd   = (t >> 4) & 3;
  out[t] = out[t] / (z[node * NH + hd] + 1e-6f);
}

extern "C" void kernel_launch(void* const* d_in, const int* in_sizes, int n_in,
                              void* d_out, int out_size, void* d_ws,
                              size_t ws_size, hipStream_t stream) {
  const float* h  = (const float*)d_in[0];
  const float* Wq = (const float*)d_in[1];
  const float* bq = (const float*)d_in[2];
  const float* Wk = (const float*)d_in[3];
  const float* bk = (const float*)d_in[4];
  const float* Wv = (const float*)d_in[5];
  const float* bv = (const float*)d_in[6];
  const int*  src = (const int*)d_in[7];
  const int*  dst = (const int*)d_in[8];

  int N = in_sizes[0] / IN_DIM;
  int E = in_sizes[7];
  float* out = (float*)d_out;

  // scratch layout: Q | K | V | z | Wpack(bf16)
  float*  Q  = (float*)d_ws;
  float*  K  = Q + (size_t)N * HD;
  float*  V  = K + (size_t)N * HD;
  float*  z  = V + (size_t)N * HD;
  bf16_t* wp = (bf16_t*)(z + (size_t)N * NH);

  int nOut = N * HD, nZ = N * NH;
  zero_kernel<<<(nOut + 255) / 256, 256, 0, stream>>>(out, z, nOut, nZ);
  pack_w_kernel<<<6, 256, 0, stream>>>(Wq, Wk, Wv, wp);

  int nTiles = (N + 15) / 16;
  qkv_gemm_kernel<<<(nTiles + 7) / 8, 256, 0, stream>>>(h, wp, bq, bk, bv, Q,
                                                        K, V, nTiles);
  int EH = E * NH;
  edge_kernel<<<(EH + 255) / 256, 256, 0, stream>>>(src, dst, Q, K, V, out, z,
                                                    EH);
  norm_kernel<<<(nOut + 255) / 256, 256, 0, stream>>>(out, z, nOut);
}
